// MoELayer_25778393711216
// MI455X (gfx1250) — compile-verified, matrix operand order checked by reading, not run on hardware
//
#include <hip/hip_runtime.h>
#include <hip/hip_bf16.h>
#include <stdint.h>

#define D_EMB 512
#define H_HID 2048
#define NEXP  8

typedef __bf16        bf16x16 __attribute__((ext_vector_type(16)));
typedef float         f32x8   __attribute__((ext_vector_type(8)));
typedef unsigned int  u32x4   __attribute__((ext_vector_type(4)));

union FragU { u32x4 q[2]; bf16x16 v; };

__device__ __forceinline__ unsigned short f2bf(float f) {
  unsigned int u = __float_as_uint(f);
  u += 0x7FFFu + ((u >> 16) & 1u);          // round-to-nearest-even
  return (unsigned short)(u >> 16);
}

// fast sigmoid: v_exp_f32 + v_rcp_f32 (both TRANS ops, co-issue with XDL)
__device__ __forceinline__ float fast_sigmoid(float g) {
  return __builtin_amdgcn_rcpf(1.0f + __expf(-g));
}

// ---------------------------------------------------------------- zero fill
__global__ void zero_f32(float* __restrict__ p, int n) {
  int i = blockIdx.x * blockDim.x + threadIdx.x;
  if (i < n) p[i] = 0.0f;
}
__global__ void zero_i32(int* __restrict__ p, int n) {
  int i = blockIdx.x * blockDim.x + threadIdx.x;
  if (i < n) p[i] = 0;
}

// ---------------------------------------------------- weight convert+swizzle
// Input:  W[e][k][n] f32 row-major, K x N per expert.
// Output: per (e, 16-col tile nt, 32-row tile kb) a 1KB block of bf16 laid out
// exactly as the v_wmma_f32_16x16x32_bf16 B-fragment:
//   chunk c (0/1), lane L: 8 values, col n = nt*16 + (L&15),
//   k = kb*32 + (L>>4)*8 + c*16 + j  (j=0..7), stored at block + c*512B + L*16B.
__global__ void wconv(const float* __restrict__ in, unsigned short* __restrict__ out,
                      int K, int N) {
  int g = blockIdx.x * blockDim.x + threadIdx.x;
  int nkb = K >> 5, nnt = N >> 4;
  int total = NEXP * nnt * nkb * 64;
  if (g >= total) return;
  int b = g >> 6, cw = g & 63;
  int c = cw >> 5, lane = cw & 31;
  int kb = b % nkb;
  int bt = b / nkb;
  int nt = bt % nnt;
  int e  = bt / nnt;
  int n = (nt << 4) + (lane & 15);
  int k0 = (kb << 5) + ((lane >> 4) << 3) + (c << 4);
  const float* src = in + (size_t)e * K * N + (size_t)k0 * N + n;
  union { unsigned short s[8]; u32x4 q; } t;
#pragma unroll
  for (int j = 0; j < 8; ++j) t.s[j] = f2bf(src[(size_t)j * N]);
  *(u32x4*)(out + ((size_t)b << 9) + ((size_t)cw << 3)) = t.q;
}

// ------------------------------------------------------------ gating / top-2
__global__ __launch_bounds__(256) void gate_topk(
    const float* __restrict__ x, const float* __restrict__ gw,
    int* __restrict__ cnt, int* __restrict__ tok, float* __restrict__ wt, int T) {
  __shared__ float sgw[D_EMB * NEXP];
  int tid = threadIdx.x;
  for (int i = tid; i < D_EMB * NEXP; i += 256) sgw[i] = gw[i];
  __syncthreads();
  int t = blockIdx.x * 256 + tid;
  if (t >= T) return;
  float acc[NEXP] = {};
  const float* xr = x + (size_t)t * D_EMB;
  for (int d = 0; d < D_EMB; ++d) {
    float xv = xr[d];
#pragma unroll
    for (int e = 0; e < NEXP; ++e) acc[e] += xv * sgw[d * NEXP + e];
  }
  int i0 = 0; float v0 = acc[0];
#pragma unroll
  for (int e = 1; e < NEXP; ++e) if (acc[e] > v0) { v0 = acc[e]; i0 = e; }
  int i1 = -1; float v1 = -3.4e38f;
#pragma unroll
  for (int e = 0; e < NEXP; ++e) if (e != i0 && acc[e] > v1) { v1 = acc[e]; i1 = e; }
  float e1 = __expf(v1 - v0);
  float p0 = __builtin_amdgcn_rcpf(1.0f + e1);   // softmax over top-2
  float p1 = e1 * p0;
  int pos0 = atomicAdd(&cnt[i0], 1);
  tok[(size_t)i0 * T + pos0] = t; wt[(size_t)i0 * T + pos0] = p0;
  int pos1 = atomicAdd(&cnt[i1], 1);
  tok[(size_t)i1 * T + pos1] = t; wt[(size_t)i1 * T + pos1] = p1;
}

// --------------------------------------------- fragment loaders (per-wave)
__device__ __forceinline__ bf16x16 load_afrag_lds(const unsigned short* base,
                                                  int stride, int row16,
                                                  int kbase, int lane) {
  FragU f;
  const unsigned short* p = base + (row16 + (lane & 15)) * stride + kbase + ((lane >> 4) << 3);
  f.q[0] = *(const u32x4*)(p);
  f.q[1] = *(const u32x4*)(p + 16);
  return f.v;
}
__device__ __forceinline__ bf16x16 load_bfrag_gbl(const unsigned short* blk, int lane) {
  FragU f;
  f.q[0] = *(const u32x4*)(blk + lane * 8);
  f.q[1] = *(const u32x4*)(blk + 256 + lane * 8);
  return f.v;
}

// ------------------------------------------------------------ expert FFN
// One block = 32-token tile of one expert's routed list.
// Fused SwiGLU (x@W1, x@W3) -> h chunk in LDS -> h@W2 accumulated in VGPRs.
__global__ __launch_bounds__(256) void moe_ffn(
    const float* __restrict__ x,
    const unsigned short* __restrict__ W1b,
    const unsigned short* __restrict__ W3b,
    const unsigned short* __restrict__ W2b,
    const int* __restrict__ cnt,
    const int* __restrict__ tokl,
    const float* __restrict__ wtl,
    float* __restrict__ out, int T) {
  __shared__ __align__(16) unsigned short sA[32 * 520];   // x tile, bf16, padded
  __shared__ __align__(16) unsigned short sH[32 * 264];   // h chunk, bf16, padded
  __shared__ int   stok[32];
  __shared__ float swt[32];

  const int e = blockIdx.y;
  const int tile = blockIdx.x;
  const int count = cnt[e];
  if (tile * 32 >= count) return;          // uniform: whole block exits

  const int tid  = threadIdx.x;
  const int lane = tid & 31;
  const int wave = tid >> 5;
  const int lcol = lane & 15;
  const int lhi8 = (lane >> 4) << 3;

  if (tid < 32) {
    int i = tile * 32 + tid;
    if (i < count) { stok[tid] = tokl[(size_t)e * T + i]; swt[tid] = wtl[(size_t)e * T + i]; }
    else           { stok[tid] = tokl[(size_t)e * T + count - 1]; swt[tid] = 0.0f; }
  }
  __syncthreads();

  // gather + convert 32 token rows (coalesced f32 reads)
  for (int i = tid; i < 32 * D_EMB; i += 256) {
    int r = i >> 9, cc = i & 511;
    sA[r * 520 + cc] = f2bf(x[(size_t)stok[r] * D_EMB + cc]);
  }
  __syncthreads();

  f32x8 accO[8];
#pragma unroll
  for (int i = 0; i < 8; ++i) accO[i] = (f32x8){0.f,0.f,0.f,0.f,0.f,0.f,0.f,0.f};

  for (int ch = 0; ch < 8; ++ch) {          // H in chunks of 256 cols
    // ---- phase A: h chunk = silu(x@W1) * (x@W3) ----
    for (int pass = 0; pass < 2; ++pass) {
      const int gnt = ch * 16 + pass * 8 + wave;        // 16-col tile index in H
      const int n0l = pass * 128 + wave * 16;           // local col in sH
#pragma unroll
      for (int half = 0; half < 2; ++half) {
        f32x8 a1 = (f32x8){0.f,0.f,0.f,0.f,0.f,0.f,0.f,0.f};
        f32x8 a3 = (f32x8){0.f,0.f,0.f,0.f,0.f,0.f,0.f,0.f};
        const int row16 = half * 16;
#pragma unroll 4
        for (int kb = 0; kb < 16; ++kb) {               // K = D = 512
          bf16x16 fa = load_afrag_lds(sA, 520, row16, kb * 32, lane);
          const size_t blk = (((size_t)e * 128 + gnt) * 16 + kb) * 512;
          bf16x16 f1 = load_bfrag_gbl(W1b + blk, lane);
          bf16x16 f3 = load_bfrag_gbl(W3b + blk, lane);
          a1 = __builtin_amdgcn_wmma_f32_16x16x32_bf16(false, fa, false, f1, (short)0, a1, false, false);
          a3 = __builtin_amdgcn_wmma_f32_16x16x32_bf16(false, fa, false, f3, (short)0, a3, false, false);
        }
#pragma unroll
        for (int r = 0; r < 8; ++r) {
          float g = a1[r];
          float hv = g * fast_sigmoid(g) * a3[r];       // silu(g) * up
          sH[(row16 + r + lhi8) * 264 + n0l + lcol] = f2bf(hv);
        }
      }
    }
    __syncthreads();
    // ---- phase B: out_tile += h_chunk @ W2[chunk rows] ----
#pragma unroll
    for (int t4 = 0; t4 < 4; ++t4) {
      const int nt2 = wave * 4 + t4;                    // 16-col tile in D
#pragma unroll
      for (int half = 0; half < 2; ++half) {
        f32x8 acc = accO[t4 * 2 + half];
        const int row16 = half * 16;
#pragma unroll 4
        for (int kb = 0; kb < 8; ++kb) {                // 256 K this chunk
          bf16x16 fa = load_afrag_lds(sH, 264, row16, kb * 32, lane);
          const size_t blk = (((size_t)e * 32 + nt2) * 64 + ch * 8 + kb) * 512;
          bf16x16 fb = load_bfrag_gbl(W2b + blk, lane);
          acc = __builtin_amdgcn_wmma_f32_16x16x32_bf16(false, fa, false, fb, (short)0, acc, false, false);
        }
        accO[t4 * 2 + half] = acc;
      }
    }
    __syncthreads();
  }

  // routing-weight scale + accumulate (each token sits in exactly 2 lists)
#pragma unroll
  for (int t4 = 0; t4 < 4; ++t4) {
#pragma unroll
    for (int half = 0; half < 2; ++half) {
#pragma unroll
      for (int r = 0; r < 8; ++r) {
        int m = half * 16 + r + lhi8;
        int n = (wave << 6) + (t4 << 4) + lcol;
        float v = accO[t4 * 2 + half][r] * swt[m];
        atomicAdd(out + (size_t)stok[m] * D_EMB + n, v);
      }
    }
  }
}

// ------------------------------------------------------------------ launch
extern "C" void kernel_launch(void* const* d_in, const int* in_sizes, int n_in,
                              void* d_out, int out_size, void* d_ws, size_t ws_size,
                              hipStream_t stream) {
  const float* x  = (const float*)d_in[0];
  const float* gw = (const float*)d_in[1];
  const float* W1 = (const float*)d_in[2];
  const float* W2 = (const float*)d_in[3];   // note dict order: W2 before W3
  const float* W3 = (const float*)d_in[4];
  float* out = (float*)d_out;
  const int T = in_sizes[0] / D_EMB;         // 32768 tokens

  char* ws = (char*)d_ws;
  const size_t wbytes = (size_t)NEXP * (D_EMB / 32) * (H_HID / 16) * 1024; // 16.78MB (W2 same)
  unsigned short* W1b = (unsigned short*)(ws);
  unsigned short* W3b = (unsigned short*)(ws + wbytes);
  unsigned short* W2b = (unsigned short*)(ws + 2 * wbytes);
  int*   cntp = (int*)(ws + 3 * wbytes);
  int*   tokp = (int*)(ws + 3 * wbytes + 1024);
  float* wtp  = (float*)(ws + 3 * wbytes + 1024 + (size_t)NEXP * T * sizeof(int));

  zero_f32<<<(out_size + 255) / 256, 256, 0, stream>>>(out, out_size);
  zero_i32<<<1, 32, 0, stream>>>(cntp, NEXP);

  int nch1 = NEXP * (H_HID / 16) * (D_EMB / 32) * 64;
  wconv<<<(nch1 + 255) / 256, 256, 0, stream>>>(W1, W1b, D_EMB, H_HID);
  wconv<<<(nch1 + 255) / 256, 256, 0, stream>>>(W3, W3b, D_EMB, H_HID);
  int nch2 = NEXP * (D_EMB / 16) * (H_HID / 32) * 64;
  wconv<<<(nch2 + 255) / 256, 256, 0, stream>>>(W2, W2b, H_HID, D_EMB);

  gate_topk<<<(T + 255) / 256, 256, 0, stream>>>(x, gw, cntp, tokp, wtp, T);

  dim3 grid((T + 31) / 32, NEXP);
  moe_ffn<<<grid, 256, 0, stream>>>(x, W1b, W3b, W2b, cntp, tokp, wtp, out, T);
}
